// NeuralNetwork_7215545057959
// MI455X (gfx1250) — matrix operand order repaired and checked by reading, not compiled, not run on hardware
//
#include <hip/hip_runtime.h>

typedef __attribute__((ext_vector_type(8)))  _Float16 v8h;
typedef __attribute__((ext_vector_type(16))) _Float16 v16h;
typedef __attribute__((ext_vector_type(8)))  float    v8f;
typedef __attribute__((ext_vector_type(4)))  unsigned int u32x4;
typedef __attribute__((ext_vector_type(8)))  int          i32x8;
typedef __attribute__((ext_vector_type(4)))  int          i32x4;

#define HIDDEN       128
#define LAYERS       9
#define ACT_STRIDE   136          // 128 + 8 halfs padding: conflict-free strided A reads
#define TPB          256          // 8 waves (wave32)
#define WAVES        8
#define PTS_PER_WAVE 32           // two 16-row M-tiles per wave -> B frags feed 2 WMMAs
#define PTS_PER_WG   (WAVES * PTS_PER_WAVE)     // 256
#define LAYER_HALFS  (HIDDEN * HIDDEN)          // 16384 f16 = 32 KB per layer
#define ACT_HALFS    (PTS_PER_WAVE * ACT_STRIDE)

#if defined(__AMDGCN__) && __has_builtin(__builtin_amdgcn_tensor_load_to_lds)
#define USE_TDM 1
#else
#define USE_TDM 0
#endif

// ---------------------------------------------------------------------------
// Pack Wh (fp32, [9][128][128], row o = out-channel, col c = in-channel) into
// f16 B-fragment order for v_wmma_f32_16x16x32_f16:
//   fragment (L, ktile k, otile t): 512 halfs; lane l holds 16 contiguous halfs
//   j=0..15 with o = t*16 + (l&15), c = k*32 + 16*(l>>4) + j
// -> each lane's B operand is one contiguous 32-byte chunk.
// ---------------------------------------------------------------------------
__global__ void pack_weights(const float* __restrict__ Wh, _Float16* __restrict__ wp) {
    int idx = blockIdx.x * blockDim.x + threadIdx.x;
    if (idx >= LAYERS * LAYER_HALFS) return;
    int L   = idx / LAYER_HALFS;
    int r   = idx % LAYER_HALFS;
    int k   = r / 4096;
    int rem = r % 4096;
    int t   = rem / 512;
    int q   = rem % 512;
    int l   = q / 16;
    int j   = q % 16;
    int o   = t * 16 + (l & 15);
    int c   = k * 32 + ((l >> 4) * 16) + j;
    wp[idx] = (_Float16)Wh[L * LAYER_HALFS + o * HIDDEN + c];
}

#if USE_TDM
// ---------------------------------------------------------------------------
// Issue one TDM descriptor: DMA a 32 KB (16384 x 2B) layer tile global->LDS.
// Per-wave instruction, tracked by TENSORcnt.
// ---------------------------------------------------------------------------
__device__ __forceinline__ void tdm_load_layer(unsigned lds_off,
                                               const _Float16* __restrict__ gsrc) {
    unsigned long long ga = (unsigned long long)(uintptr_t)gsrc;
    u32x4 g0;
    g0[0] = 1u;                                                  // count=1, user mode
    g0[1] = lds_off;                                             // LDS byte address
    g0[2] = (unsigned)ga;                                        // global_addr[31:0]
    g0[3] = (unsigned)((ga >> 32) & 0x01FFFFFFu) | (2u << 30);   // [56:32] | type=2
    i32x8 g1;
    g1[0] = (int)(1u << 16);          // workgroup_mask=0, data_size=1 (2-byte elems)
    g1[1] = (int)(0x4000u << 16);     // tensor_dim0 lo16 = 16384
    g1[2] = (int)(1u << 16);          // tensor_dim0 hi16 = 0 | tensor_dim1 lo16 = 1
    g1[3] = (int)(0x4000u << 16);     // tensor_dim1 hi16 = 0 | tile_dim0 = 16384
    g1[4] = 1;                        // tile_dim1 = 1, tile_dim2 = 0
    g1[5] = 16384;                    // tensor_dim0_stride lo32
    g1[6] = (int)(0x4000u << 16);     // stride0 hi16 = 0 | tensor_dim1_stride lo16
    g1[7] = 0;
    i32x4 z4 = {0, 0, 0, 0};
#if __clang_major__ >= 23
    i32x8 z8 = {0, 0, 0, 0, 0, 0, 0, 0};
    __builtin_amdgcn_tensor_load_to_lds(g0, g1, z4, z4, z8, 0);
#else
    __builtin_amdgcn_tensor_load_to_lds(g0, g1, z4, z4, 0);
#endif
}
#endif

// ---------------------------------------------------------------------------
// Fused MLP: x[N,4] -> relu(W0) -> 9x relu(Wh) via WMMA -> Wout -> out[N]
// ---------------------------------------------------------------------------
__global__ void __launch_bounds__(TPB)
mlp_fused(const float4* __restrict__ x,      // [N] of float4
          const float4* __restrict__ W0,     // [128] of float4
          const float*  __restrict__ b0,     // [128]
          const _Float16* __restrict__ wp,   // packed weights, 9*16384 halfs
          const float*  __restrict__ bh,     // [9][128]
          const float*  __restrict__ wout,   // [128]
          const float*  __restrict__ boutp,  // [1]
          float* __restrict__ out,           // [N]
          int nPts, int nTiles)
{
    extern __shared__ char smem[];
    _Float16* sW   = (_Float16*)smem;                             // 2 x 16384 halfs
    _Float16* sAct = (_Float16*)(smem + 2 * LAYER_HALFS * 2);     // 8 x 32*136 halfs

    const int tid  = threadIdx.x;
    const int wave = tid >> 5;
    const int lane = tid & 31;
    const int nlo  = lane & 15;      // A-layout M / D-layout N
    const int g16  = lane >> 4;      // lane half-group
    _Float16* actW = sAct + wave * ACT_HALFS;   // wave-private activations (32 pts)

    const float bout = boutp[0];

    // A-fragment loader: lane holds K run [16g,16g+16) of ktile k
    const _Float16* aBase = actW + nlo * ACT_STRIDE + g16 * 16;
    auto loadA = [&](int k, v16h& A0, v16h& A1) {
        const _Float16* r0 = aBase + k * 32;
        const _Float16* r1 = r0 + 16 * ACT_STRIDE;
        v8h lo0 = *(const v8h*)(r0);
        v8h hi0 = *(const v8h*)(r0 + 8);
        v8h lo1 = *(const v8h*)(r1);
        v8h hi1 = *(const v8h*)(r1 + 8);
        A0 = __builtin_shufflevector(lo0, hi0, 0,1,2,3,4,5,6,7,8,9,10,11,12,13,14,15);
        A1 = __builtin_shufflevector(lo1, hi1, 0,1,2,3,4,5,6,7,8,9,10,11,12,13,14,15);
    };

    for (int tIdx = blockIdx.x; tIdx < nTiles; tIdx += gridDim.x) {
        const int basePt = tIdx * PTS_PER_WG + wave * PTS_PER_WAVE;

        // ---- stage layer-0 weights (latency hidden by the input layer)
#if USE_TDM
        if (wave == 0)
            tdm_load_layer((unsigned)(uintptr_t)sW, wp);
#else
        uint4 wreg[8];
        {
            const uint4* gsrc = (const uint4*)wp;
#pragma unroll
            for (int i = 0; i < 8; ++i) wreg[i] = gsrc[tid + i * TPB];
        }
#endif

        // ---- input layer: h0[p][c] = relu(x[p][0..3] . W0[c][0..3] + b0[c])
        {
            float4 w0c[4]; float b0c[4];
#pragma unroll
            for (int cb = 0; cb < 4; ++cb) {
                w0c[cb] = W0[lane + cb * 32];
                b0c[cb] = b0[lane + cb * 32];
            }
#pragma unroll 4
            for (int p = 0; p < PTS_PER_WAVE; ++p) {
                float4 xv = (basePt + p < nPts) ? x[basePt + p]
                                                : make_float4(0.f, 0.f, 0.f, 0.f);
#pragma unroll
                for (int cb = 0; cb < 4; ++cb) {
                    float v = fmaf(xv.x, w0c[cb].x,
                              fmaf(xv.y, w0c[cb].y,
                              fmaf(xv.z, w0c[cb].z,
                              fmaf(xv.w, w0c[cb].w, b0c[cb]))));
                    actW[p * ACT_STRIDE + lane + cb * 32] = (_Float16)fmaxf(v, 0.0f);
                }
            }
        }

        // ---- publish layer-0 weights
#if USE_TDM
        if (wave == 0) __builtin_amdgcn_s_wait_tensorcnt(0);
#else
        {
            uint4* sdst = (uint4*)sW;
#pragma unroll
            for (int i = 0; i < 8; ++i) sdst[tid + i * TPB] = wreg[i];
        }
#endif
        __syncthreads();

        // ---- 9 hidden layers, double-buffered weights, 1 barrier/layer
        for (int layer = 0; layer < LAYERS; ++layer) {
            const _Float16* sWr = sW + (layer & 1) * LAYER_HALFS;

            // kick off next layer's weight DMA while this layer's WMMAs run
            if (layer + 1 < LAYERS) {
#if USE_TDM
                if (wave == 0)
                    tdm_load_layer(
                        (unsigned)(uintptr_t)(sW + ((layer + 1) & 1) * LAYER_HALFS),
                        wp + (layer + 1) * LAYER_HALFS);
#else
                const uint4* gsrc = (const uint4*)(wp + (layer + 1) * LAYER_HALFS);
#pragma unroll
                for (int i = 0; i < 8; ++i) wreg[i] = gsrc[tid + i * TPB];
#endif
            }

            // bias-initialized accumulators for both 16-row M-tiles
            v8f acc[2][8];
#pragma unroll
            for (int t = 0; t < 8; ++t) {
                float bv = bh[layer * HIDDEN + t * 16 + nlo];
#pragma unroll
                for (int r = 0; r < 8; ++r) { acc[0][t][r] = bv; acc[1][t][r] = bv; }
            }

            // software-pipelined k/t loops: next A and next B issue before the
            // WMMAs consuming the current ones -> ds loads get a full WMMA-pair
            // of latency cover instead of s_wait_dscnt 0 stalls.
            v16h a0c, a1c, a0n, a1n;
            loadA(0, a0c, a1c);
            const v16h* bAll = (const v16h*)sWr;
#pragma unroll
            for (int k = 0; k < 4; ++k) {
                if (k < 3) loadA(k + 1, a0n, a1n);
                const v16h* bfr = bAll + k * 256 + lane;
                v16h bc = bfr[0];
#pragma unroll
                for (int t = 0; t < 8; ++t) {
                    v16h bn = (t < 7) ? bfr[(t + 1) * 32] : bc;
                    acc[0][t] = __builtin_amdgcn_wmma_f32_16x16x32_f16(
                        false, a0c, false, bc, (short)0, acc[0][t], false, false);
                    acc[1][t] = __builtin_amdgcn_wmma_f32_16x16x32_f16(
                        false, a1c, false, bc, (short)0, acc[1][t], false, false);
                    bc = bn;
                }
                a0c = a0n; a1c = a1n;
            }

#if !USE_TDM
            // commit next layer's weights to the other LDS buffer
            if (layer + 1 < LAYERS) {
                uint4* sdst = (uint4*)(sW + ((layer + 1) & 1) * LAYER_HALFS);
#pragma unroll
                for (int i = 0; i < 8; ++i) sdst[tid + i * TPB] = wreg[i];
            }
#endif

            // relu + write back activations (wave-private region)
            // D layout: elem r -> point r + 8*g16 (+16 for 2nd M-tile); lane -> ch t*16+nlo
#pragma unroll
            for (int g = 0; g < 2; ++g) {
#pragma unroll
                for (int t = 0; t < 8; ++t) {
#pragma unroll
                    for (int r = 0; r < 8; ++r) {
                        int p = r + g16 * 8 + g * 16;
                        actW[p * ACT_STRIDE + t * 16 + nlo] =
                            (_Float16)fmaxf(acc[g][t][r], 0.0f);
                    }
                }
            }

#if USE_TDM
            // drain this wave's outstanding weight DMA before publishing the buffer
            if (layer + 1 < LAYERS && wave == 0)
                __builtin_amdgcn_s_wait_tensorcnt(0);
#endif
            __syncthreads();
        }

        // ---- output layer: one point per lane, vectorized activation reads
        {
            const int p = lane;
            float s = bout;
#pragma unroll
            for (int cb = 0; cb < 16; ++cb) {
                v8h av = *(const v8h*)(actW + p * ACT_STRIDE + cb * 8);
#pragma unroll
                for (int j = 0; j < 8; ++j)
                    s = fmaf((float)av[j], wout[cb * 8 + j], s);
            }
            if (basePt + p < nPts) out[basePt + p] = s;
        }
    }
}

// ---------------------------------------------------------------------------
extern "C" void kernel_launch(void* const* d_in, const int* in_sizes, int n_in,
                              void* d_out, int out_size, void* d_ws, size_t ws_size,
                              hipStream_t stream) {
    const float* x    = (const float*)d_in[0];
    const float* W0   = (const float*)d_in[1];
    const float* b0   = (const float*)d_in[2];
    const float* Wh   = (const float*)d_in[3];
    const float* bh   = (const float*)d_in[4];
    const float* Wout = (const float*)d_in[5];
    const float* bout = (const float*)d_in[6];
    float* out = (float*)d_out;

    _Float16* wp = (_Float16*)d_ws;   // 9*16384 halfs = 288 KB of scratch

    const int packTotal = LAYERS * LAYER_HALFS;
    pack_weights<<<(packTotal + TPB - 1) / TPB, TPB, 0, stream>>>(Wh, wp);

    const int nPts   = in_sizes[0] / 4;
    const int nTiles = (nPts + PTS_PER_WG - 1) / PTS_PER_WG;   // 4096 for N=1M
    const int blocks = nTiles < 1024 ? nTiles : 1024;          // grid-stride persistence
    const size_t smemBytes = (size_t)2 * LAYER_HALFS * 2        // weights (dbl-buf): 64 KB
                           + (size_t)WAVES * ACT_HALFS * 2;     // acts: 68 KB

    mlp_fused<<<blocks, TPB, smemBytes, stream>>>(
        (const float4*)x, (const float4*)W0, b0, wp, bh, Wout, bout, out,
        nPts, nTiles);
}
